// SequenceModelFixedLen_62818191671710
// MI455X (gfx1250) — compile-verified, matrix operand order checked by reading, not run on hardware
//
#include <hip/hip_runtime.h>
#include <hip/hip_bf16.h>
#include <math.h>

// ---------------------------------------------------------------------------
// RNN with per-timestep weights on gfx1250 (MI455X).
//   h_{t+1} = tanh(x_t @ W_xh[t]^T + h_t @ W_hh[t]^T + b_hh[t]), frozen at
//   t >= seq_len[b];  out = h_last @ W_out^T + b_out.
//
// v2 changes vs v1:
//   * hardware tanh (V_TANH_F32) via __builtin_amdgcn_tanhf — kills the ~1.5k
//     VALU/step ocml expansion that dominated the round-1 histogram.
//   * weight staging via Tensor Data Mover (tensor_load_to_lds) with a 1-D D#
//     (12416 x 4B), double-buffered: TDM for step t+1 runs under the WMMAs of
//     step t; sync = s_wait_tensorcnt(1) + workgroup barrier. Manual-copy
//     fallback if the builtin is unavailable.
//   * per-step bias folded into the staged record (49152B weights + 512B bias).
//
// d_ws requirement: T * 49664 bytes = 24.25 MB.
// ---------------------------------------------------------------------------

typedef __attribute__((ext_vector_type(16))) _Float16 v16h;
typedef __attribute__((ext_vector_type(8)))  _Float16 v8h;
typedef __attribute__((ext_vector_type(8)))  float    v8f;
typedef __attribute__((ext_vector_type(4)))  float    v4f;
typedef __attribute__((ext_vector_type(4)))  unsigned int u32x4;
typedef __attribute__((ext_vector_type(8)))  int      i32x8;
typedef __attribute__((ext_vector_type(4)))  int      i32x4;

#define B_   2048
#define T_   512
#define I_   64
#define H_   128
#define KTOT 192                  // I_ + H_
#define NKB  6                    // K blocks of 32
#define WPT  (KTOT * H_)          // 24576 f16 weight elems per timestep
#define WREC_BYTES 49664          // 48KB weights + 512B f32 bias per timestep
#define WREC_HALFS (WREC_BYTES / 2)
#define WREC_ELEMS (WREC_BYTES / 4)   // 12416 4-byte elements for the TDM D#
#define TILE_HALFS 512            // one (kb,ntile) B-fragment tile
#define HSTRIDE 136               // padded halfs per h row -> 272B row stride
#define NW   4                    // waves per workgroup
#define THREADS (NW * 32)

union FragA { v16h v; v8h h[2]; };

// ----------------------------- tanh selection ------------------------------
#if __has_builtin(__builtin_amdgcn_tanhf)
#define TANHF(x) __builtin_amdgcn_tanhf(x)
#else
#define TANHF(x) tanhf(x)
#endif

// ------------------------------ TDM support --------------------------------
#if __has_builtin(__builtin_amdgcn_tensor_load_to_lds)
#define HAVE_TDM 1
#else
#define HAVE_TDM 0
#endif

#if __has_builtin(__builtin_amdgcn_s_wait_tensorcnt)
#define TENSOR_WAIT(n) __builtin_amdgcn_s_wait_tensorcnt(n)
#else
#define TENSOR_WAIT(n) asm volatile("s_wait_tensorcnt %0" ::"i"(n) : "memory")
#endif

#if HAVE_TDM
// 1-D DMA: WREC_BYTES bytes from global 'gptr' into LDS byte offset 'ldsaddr'.
// D# per CDNA5 ISA ch.8: group0 = {count=1 | lds_addr | global_addr | type=2},
// group1 = {data_size=2(4B) | tensor_dim0=tile_dim0=12416 | dim0_stride=12416}.
__device__ __forceinline__ void tdm_load_rec(unsigned int ldsaddr, const void* gptr) {
    unsigned long long ga = (unsigned long long)(uintptr_t)gptr;
    u32x4 g0;
    g0[0] = 1u;                                         // count=1, user mode
    g0[1] = ldsaddr;                                    // LDS byte address
    g0[2] = (unsigned int)(ga & 0xffffffffu);           // global_addr[31:0]
    g0[3] = (unsigned int)((ga >> 32) & 0x01ffffffu)    // global_addr[56:32]
          | (2u << 30);                                 // type=2 ("image")
    i32x8 g1;
    g1[0] = (int)(2u << 16);                            // data_size=2 -> 4B
    g1[1] = (int)((WREC_ELEMS & 0xffff) << 16);         // tensor_dim0[15:0]
    g1[2] = (int)((WREC_ELEMS >> 16) | (1 << 16));      // dim0 hi | tensor_dim1=1
    g1[3] = (int)(WREC_ELEMS << 16);                    // tile_dim0
    g1[4] = 0;                                          // tile_dim1/2 unused
    g1[5] = (int)WREC_ELEMS;                            // tensor_dim0_stride lo
    g1[6] = 0;
    g1[7] = 0;
    i32x4 z4 = {0, 0, 0, 0};
#if __clang_major__ >= 23
    i32x8 z8 = {0, 0, 0, 0, 0, 0, 0, 0};
    __builtin_amdgcn_tensor_load_to_lds(g0, g1, z4, z4, z8, 0);
#else
    __builtin_amdgcn_tensor_load_to_lds(g0, g1, z4, z4, 0);
#endif
}
#endif

// -------------------------- weight repack kernels --------------------------
// Per-t record (bytes): [0,49152) f16 weights in WMMA B-fragment order,
// [49152,49664) f32 bias. Fragment order (halfs): (kb*8+ntile)*512 +
// chunk*256 + lane*8 + e  holds  W[n][k],  n = ntile*16 + (lane&15),
// k = kb*32 + ((lane>>4)&1)*8 + chunk*16 + e;  k<64 -> W_xh, else W_hh.
__global__ __launch_bounds__(256) void prep_weights(
    const float* __restrict__ Wxh, const float* __restrict__ Whh,
    unsigned char* __restrict__ ws)
{
    int idx = blockIdx.x * 256 + threadIdx.x;          // < T_*WPT
    if (idx >= T_ * WPT) return;
    int t     = idx / WPT;
    int r     = idx % WPT;
    int blk   = r >> 9;
    int r2    = r & 511;
    int chunk = r2 >> 8;
    int r3    = r2 & 255;
    int lane  = r3 >> 3;
    int e     = r3 & 7;
    int kb    = blk >> 3;
    int ntile = blk & 7;
    int n = ntile * 16 + (lane & 15);
    int k = kb * 32 + ((lane >> 4) & 1) * 8 + chunk * 16 + e;
    float v = (k < I_) ? Wxh[((size_t)t * H_ + n) * I_ + k]
                       : Whh[((size_t)t * H_ + n) * H_ + (k - I_)];
    ((_Float16*)(ws + (size_t)t * WREC_BYTES))[r] = (_Float16)v;
}

__global__ __launch_bounds__(256) void prep_bias(
    const float* __restrict__ bhh, unsigned char* __restrict__ ws)
{
    int idx = blockIdx.x * 256 + threadIdx.x;          // < T_*H_
    if (idx >= T_ * H_) return;
    int t = idx >> 7;
    int n = idx & 127;
    ((float*)(ws + (size_t)t * WREC_BYTES + WPT * 2))[n] = bhh[idx];
}

// ------------------------------ main kernel --------------------------------
__global__ __launch_bounds__(THREADS) void rnn_main(
    const float* __restrict__ x,         // [B,T,I]
    const int*   __restrict__ lens,      // [B]
    const unsigned char* __restrict__ ws,// repacked [T][WREC_BYTES]
    const float* __restrict__ wout,      // [O=1,H]
    const float* __restrict__ bout,      // [1]
    float* __restrict__ out)             // [B,1]
{
    __shared__ __align__(16) unsigned char wstage[2][WREC_BYTES]; // 97 KB
    __shared__ __align__(16) _Float16 hlds[NW * 16 * HSTRIDE];    // 17 KB

    const int tid  = threadIdx.x;
    const int wave = tid >> 5;
    const int lane = tid & 31;
    const int l15  = lane & 15;
    const int hi8  = (lane >> 4) << 3;                 // 0 or 8
    const int bbase = blockIdx.x * (NW * 16) + wave * 16;
    _Float16* hwave = hlds + wave * 16 * HSTRIDE;

    // zero recurrent state (covers seq_len==0 rows too)
    for (int j = lane; j < 16 * HSTRIDE; j += 32) hwave[j] = (_Float16)0.f;

    int mylen[8];
    #pragma unroll
    for (int r = 0; r < 8; ++r) mylen[r] = lens[bbase + r + hi8];

#if HAVE_TDM
    // prime buffer 0 with step 0's record
    if (wave == 0) tdm_load_rec((unsigned int)(uintptr_t)&wstage[0][0], ws);
#else
    {   // fallback: cooperative copy of step 0
        const v8h* src = (const v8h*)ws;
        v8h* dst = (v8h*)&wstage[0][0];
        for (int j = tid; j < WREC_BYTES / 16; j += THREADS) dst[j] = src[j];
    }
#endif

    for (int t = 0; t < T_; ++t) {
        const int cur = t & 1;
        __syncthreads();   // all waves done with buffer cur^1 (step t-1)
#if HAVE_TDM
        if (wave == 0) {
            if (t + 1 < T_) {
                tdm_load_rec((unsigned int)(uintptr_t)&wstage[cur ^ 1][0],
                             ws + (size_t)(t + 1) * WREC_BYTES);
                TENSOR_WAIT(1);   // in-order: ensures step-t load is complete
            } else {
                TENSOR_WAIT(0);
            }
        }
#else
        if (t + 1 < T_) {
            const v8h* src = (const v8h*)(ws + (size_t)(t + 1) * WREC_BYTES);
            v8h* dst = (v8h*)&wstage[cur ^ 1][0];
            for (int j = tid; j < WREC_BYTES / 16; j += THREADS)
                dst[j] = src[j];
        }
#endif
        __syncthreads();   // step-t record visible to all waves

        const _Float16* wcur = (const _Float16*)&wstage[cur][0];

        v8f acc[8];
        #pragma unroll
        for (int nt = 0; nt < 8; ++nt) {
            v8f z = {0.f, 0.f, 0.f, 0.f, 0.f, 0.f, 0.f, 0.f};
            acc[nt] = z;
        }

        const float* xrow = x + ((size_t)(bbase + l15) * T_ + t) * I_;
        #pragma unroll
        for (int kb = 0; kb < NKB; ++kb) {
            FragA a;
            if (kb < 2) {
                // A-fragment from x: f32 loads + packed cvt to f16
                #pragma unroll
                for (int c = 0; c < 2; ++c) {
                    const float* p = xrow + kb * 32 + hi8 + c * 16;
                    v4f f0 = *(const v4f*)(p);
                    v4f f1 = *(const v4f*)(p + 4);
                    v8h hh;
                    #pragma unroll
                    for (int e = 0; e < 4; ++e) {
                        hh[e]     = (_Float16)f0[e];
                        hh[e + 4] = (_Float16)f1[e];
                    }
                    a.h[c] = hh;
                }
            } else {
                // A-fragment from recurrent h (f16 LDS, conflict-free layout)
                const _Float16* hr = hwave + l15 * HSTRIDE + (kb - 2) * 32 + hi8;
                a.h[0] = *(const v8h*)(hr);
                a.h[1] = *(const v8h*)(hr + 16);
            }
            #pragma unroll
            for (int nt = 0; nt < 8; ++nt) {
                const _Float16* wt = wcur + (kb * 8 + nt) * TILE_HALFS + lane * 8;
                FragA bfr;
                bfr.h[0] = *(const v8h*)(wt);
                bfr.h[1] = *(const v8h*)(wt + 256);
                acc[nt] = __builtin_amdgcn_wmma_f32_16x16x32_f16(
                    false, a.v, false, bfr.v, (short)0, acc[nt], false, false);
            }
        }

        // epilogue: bias (staged in LDS) + hardware tanh + freeze-mask
        const float* bb = (const float*)(wstage[cur] + WPT * 2);
        #pragma unroll
        for (int nt = 0; nt < 8; ++nt) {
            float bias = bb[nt * 16 + l15];
            #pragma unroll
            for (int r = 0; r < 8; ++r) {
                float v = TANHF(acc[nt][r] + bias);
                if (t < mylen[r])   // skip store => hidden stays frozen
                    hwave[(r + hi8) * HSTRIDE + nt * 16 + l15] = (_Float16)v;
            }
        }
    }

    __syncthreads();
    // final projection H -> O=1: lanes 0..15 each handle one batch row
    if (lane < 16) {
        const _Float16* hr = hwave + lane * HSTRIDE;
        float s = 0.f;
        for (int j = 0; j < H_; ++j) s += (float)hr[j] * wout[j];
        out[bbase + lane] = s + bout[0];
    }
}

// ------------------------------- launcher ----------------------------------
extern "C" void kernel_launch(void* const* d_in, const int* in_sizes, int n_in,
                              void* d_out, int out_size, void* d_ws, size_t ws_size,
                              hipStream_t stream) {
    const float* x    = (const float*)d_in[0];
    const int*   lens = (const int*)d_in[1];
    const float* Wxh  = (const float*)d_in[2];
    const float* Whh  = (const float*)d_in[3];
    const float* bhh  = (const float*)d_in[4];
    const float* Wout = (const float*)d_in[5];
    const float* bout = (const float*)d_in[6];
    float* outp = (float*)d_out;
    unsigned char* ws = (unsigned char*)d_ws;   // needs T_*WREC_BYTES = 24.25MB

    int totalW = T_ * WPT;
    prep_weights<<<(totalW + 255) / 256, 256, 0, stream>>>(Wxh, Whh, ws);
    int totalB = T_ * H_;
    prep_bias<<<(totalB + 255) / 256, 256, 0, stream>>>(bhh, ws);
    rnn_main<<<B_ / (NW * 16), THREADS, 0, stream>>>(
        x, lens, ws, Wout, bout, outp);
}
